// PositionBasedSummaryForMLM_1125281432112
// MI455X (gfx1250) — matrix operand order repaired and checked by reading, not compiled, not run on hardware
//
#include <hip/hip_runtime.h>
#include <hip/hip_bf16.h>
#include <math.h>

typedef __bf16 bf16_t;
typedef __attribute__((ext_vector_type(8)))  __bf16 v8bf;
typedef __attribute__((ext_vector_type(16))) __bf16 v16bf;
typedef __attribute__((ext_vector_type(8)))  float  v8f;

#define WAVES_PER_BLOCK 4
#define TM 2 /* 2 x 16 = 32 output rows per wave */
#define TN 4 /* 4 x 16 = 64 output columns per wave */

#define SHUF16(lo, hi) \
    __builtin_shufflevector(lo, hi, 0, 1, 2, 3, 4, 5, 6, 7, 8, 9, 10, 11, 12, 13, 14, 15)

enum { EPI_NONE = 0, EPI_SCORES = 1, EPI_GELU = 2, EPI_BIAS = 3 };

__device__ __forceinline__ float4 ld4(const float* p) { return *(const float4*)p; }

__device__ __forceinline__ float wave_max(float v) {
    #pragma unroll
    for (int o = 16; o > 0; o >>= 1) v = fmaxf(v, __shfl_xor(v, o, 32));
    return v;
}
__device__ __forceinline__ float wave_sum(float v) {
    #pragma unroll
    for (int o = 16; o > 0; o >>= 1) v += __shfl_xor(v, o, 32);
    return v;
}

// ---------------------------------------------------------------------------
// f32 -> bf16 bulk convert (n must be a multiple of 8)
// ---------------------------------------------------------------------------
__global__ __launch_bounds__(256) void cvt_bf16_kernel(
    const float* __restrict__ in, bf16_t* __restrict__ out, long n) {
    long i = ((long)blockIdx.x * blockDim.x + threadIdx.x) * 8;
    if (i >= n) return;
    float4 a = ld4(in + i), b = ld4(in + i + 4);
    v8bf o;
    o[0] = (bf16_t)a.x; o[1] = (bf16_t)a.y; o[2] = (bf16_t)a.z; o[3] = (bf16_t)a.w;
    o[4] = (bf16_t)b.x; o[5] = (bf16_t)b.y; o[6] = (bf16_t)b.z; o[7] = (bf16_t)b.w;
    *(v8bf*)(out + i) = o;
}

// ---------------------------------------------------------------------------
// seq_hidden prep: emit bf16 copy [b][s][h] AND bf16 transpose [b][h][s]
// (LDS 32x33 tile transpose, block 32x8, grid (768/32, 512/32, 8))
// ---------------------------------------------------------------------------
__global__ __launch_bounds__(256) void prep_seq_hidden(
    const float* __restrict__ in, bf16_t* __restrict__ outN,
    bf16_t* __restrict__ outT) {
    __shared__ float tile[32][33];
    const int b = blockIdx.z;
    const int h0 = blockIdx.x * 32, s0 = blockIdx.y * 32;
    const int tx = threadIdx.x, ty = threadIdx.y;
    const float* src = in + (long)b * 512 * 768;
    bf16_t* dN = outN + (long)b * 512 * 768;
    bf16_t* dT = outT + (long)b * 768 * 512;
    #pragma unroll
    for (int i = 0; i < 32; i += 8) {
        float v = src[(long)(s0 + ty + i) * 768 + h0 + tx];
        tile[ty + i][tx] = v;
        dN[(long)(s0 + ty + i) * 768 + h0 + tx] = (bf16_t)v;
    }
    __syncthreads();
    #pragma unroll
    for (int i = 0; i < 32; i += 8)
        dT[(long)(h0 + ty + i) * 512 + s0 + tx] = (bf16_t)tile[tx][ty + i];
}

// ---------------------------------------------------------------------------
// pos_h[l,h] = pos_emb[l,:200] . Wp[h,:200] + bp[h]  -> bf16 (K=200, VALU)
// ---------------------------------------------------------------------------
__global__ __launch_bounds__(256) void posh_kernel(
    const float* __restrict__ pos_emb, const float* __restrict__ Wp,
    const float* __restrict__ bp, bf16_t* __restrict__ pos_h) {
    int idx = blockIdx.x * blockDim.x + threadIdx.x;
    if (idx >= 512 * 768) return;
    int l = idx / 768, h = idx % 768;
    const float* pe = pos_emb + (long)l * 200;
    const float* w  = Wp + (long)h * 200;
    float s = bp[h];
    #pragma unroll 5
    for (int p = 0; p < 200; p += 4) {
        float4 x = ld4(pe + p), y = ld4(w + p);
        s = fmaf(x.x, y.x, fmaf(x.y, y.y, fmaf(x.z, y.z, fmaf(x.w, y.w, s))));
    }
    pos_h[idx] = (bf16_t)s;
}

// ---------------------------------------------------------------------------
// bf16 WMMA GEMM:  C[M,N] = A[M,K] * W[N,K]^T  (+ epilogue)
// One wave owns a 32x64 tile: 8x v_wmma_f32_16x16x32_bf16 per 32-K step,
// 12 b128 loads per 8 WMMAs (B fragments reused across 2 row-fragments).
// Fast path (full tile) has no branches in the k-loop. M must be mult of 32.
// ---------------------------------------------------------------------------
template <int EPI, bool OUT_BF16>
__global__ __launch_bounds__(128) void gemm_bf16(
    const bf16_t* __restrict__ A, long strideA,
    const bf16_t* __restrict__ B, long strideB,
    void* __restrict__ Cv, long strideC,          // elements
    const float* __restrict__ X, long strideX,    // SCORES: src[M,N]; GELU/BIAS: bias[N]
    const float* __restrict__ Mask, long strideMask,
    int M, int N, int K) {
    const int b = blockIdx.y;
    A += (long)b * strideA;
    B += (long)b * strideB;
    if (X)    X    += (long)b * strideX;
    if (Mask) Mask += (long)b * strideMask;
    float*  Cf = OUT_BF16 ? nullptr : ((float*)Cv + (long)b * strideC);
    bf16_t* Cb = OUT_BF16 ? ((bf16_t*)Cv + (long)b * strideC) : nullptr;

    const int lane = threadIdx.x & 31;
    const int wave = threadIdx.x >> 5;
    const int r  = lane & 15;
    const int hi = lane >> 4;

    const int ntiles = (N + 63) >> 6;
    const int tile = blockIdx.x * WAVES_PER_BLOCK + wave;
    if (tile >= (M >> 5) * ntiles) return;
    const int m0 = (tile / ntiles) * 32;
    const int n0 = (tile % ntiles) * 64;

    v8f acc[TM][TN];
    #pragma unroll
    for (int u = 0; u < TM; ++u)
        #pragma unroll
        for (int t = 0; t < TN; ++t)
            #pragma unroll
            for (int v = 0; v < 8; ++v) acc[u][t][v] = 0.0f;

    // A 16x32 fragment: lane half -> K in {hi*16..+15}
    const bf16_t* arow0 = A + (long)(m0 + r) * K + hi * 16;
    const bf16_t* arow1 = arow0 + 16l * K;

    if (n0 + 64 <= N) {
        const bf16_t* b0p = B + (long)(n0 +  0 + r) * K + hi * 16;
        const bf16_t* b1p = B + (long)(n0 + 16 + r) * K + hi * 16;
        const bf16_t* b2p = B + (long)(n0 + 32 + r) * K + hi * 16;
        const bf16_t* b3p = B + (long)(n0 + 48 + r) * K + hi * 16;
        for (int k0 = 0; k0 < K; k0 += 32) {
            v16bf a0 = SHUF16(*(const v8bf*)(arow0 + k0), *(const v8bf*)(arow0 + k0 + 8));
            v16bf a1 = SHUF16(*(const v8bf*)(arow1 + k0), *(const v8bf*)(arow1 + k0 + 8));
            v16bf w0 = SHUF16(*(const v8bf*)(b0p + k0),  *(const v8bf*)(b0p + k0 + 8));
            v16bf w1 = SHUF16(*(const v8bf*)(b1p + k0),  *(const v8bf*)(b1p + k0 + 8));
            v16bf w2 = SHUF16(*(const v8bf*)(b2p + k0),  *(const v8bf*)(b2p + k0 + 8));
            v16bf w3 = SHUF16(*(const v8bf*)(b3p + k0),  *(const v8bf*)(b3p + k0 + 8));
            acc[0][0] = __builtin_amdgcn_wmma_f32_16x16x32_bf16(false, a0, false, w0, (short)0, acc[0][0], false, false);
            acc[0][1] = __builtin_amdgcn_wmma_f32_16x16x32_bf16(false, a0, false, w1, (short)0, acc[0][1], false, false);
            acc[0][2] = __builtin_amdgcn_wmma_f32_16x16x32_bf16(false, a0, false, w2, (short)0, acc[0][2], false, false);
            acc[0][3] = __builtin_amdgcn_wmma_f32_16x16x32_bf16(false, a0, false, w3, (short)0, acc[0][3], false, false);
            acc[1][0] = __builtin_amdgcn_wmma_f32_16x16x32_bf16(false, a1, false, w0, (short)0, acc[1][0], false, false);
            acc[1][1] = __builtin_amdgcn_wmma_f32_16x16x32_bf16(false, a1, false, w1, (short)0, acc[1][1], false, false);
            acc[1][2] = __builtin_amdgcn_wmma_f32_16x16x32_bf16(false, a1, false, w2, (short)0, acc[1][2], false, false);
            acc[1][3] = __builtin_amdgcn_wmma_f32_16x16x32_bf16(false, a1, false, w3, (short)0, acc[1][3], false, false);
        }
    } else {
        const bf16_t* bptr[TN];
        #pragma unroll
        for (int t = 0; t < TN; ++t) {
            const int col = n0 + t * 16 + r;
            bptr[t] = (col < N) ? (B + (long)col * K + hi * 16) : nullptr;
        }
        v16bf zbf;
        #pragma unroll
        for (int j = 0; j < 16; ++j) zbf[j] = (bf16_t)0.0f;
        for (int k0 = 0; k0 < K; k0 += 32) {
            v16bf a0 = SHUF16(*(const v8bf*)(arow0 + k0), *(const v8bf*)(arow0 + k0 + 8));
            v16bf a1 = SHUF16(*(const v8bf*)(arow1 + k0), *(const v8bf*)(arow1 + k0 + 8));
            #pragma unroll
            for (int t = 0; t < TN; ++t) {
                v16bf bb;
                if (bptr[t]) bb = SHUF16(*(const v8bf*)(bptr[t] + k0), *(const v8bf*)(bptr[t] + k0 + 8));
                else         bb = zbf;
                acc[0][t] = __builtin_amdgcn_wmma_f32_16x16x32_bf16(false, a0, false, bb, (short)0, acc[0][t], false, false);
                acc[1][t] = __builtin_amdgcn_wmma_f32_16x16x32_bf16(false, a1, false, bb, (short)0, acc[1][t], false, false);
            }
        }
    }

    // Epilogue: acc[u][t][v] = C[m0 + u*16 + hi*8 + v][n0 + t*16 + r]
    #pragma unroll
    for (int t = 0; t < TN; ++t) {
        const int col = n0 + t * 16 + r;
        if (col >= N) continue;
        float mterm = 0.0f, bias = 0.0f;
        if (EPI == EPI_SCORES) mterm = Mask[col] * -10000.0f;
        if (EPI == EPI_GELU || EPI == EPI_BIAS) bias = X[col];
        #pragma unroll
        for (int u = 0; u < TM; ++u) {
            #pragma unroll
            for (int v = 0; v < 8; ++v) {
                const int row = m0 + u * 16 + hi * 8 + v;
                float val = acc[u][t][v];
                if (EPI == EPI_SCORES) {
                    val += X[(long)row * N + col] + mterm;
                } else if (EPI == EPI_GELU) {
                    val += bias;
                    val = 0.5f * val * (1.0f + erff(val * 0.70710678118654752f));
                } else if (EPI == EPI_BIAS) {
                    val += bias;
                }
                if (OUT_BF16) Cb[(long)row * N + col] = (bf16_t)val;
                else          Cf[(long)row * N + col] = val;
            }
        }
    }
}

// ---------------------------------------------------------------------------
// Row softmax (512 wide, wave per row): f32 in -> bf16 out
// ---------------------------------------------------------------------------
__global__ __launch_bounds__(128) void softmax512(
    const float* __restrict__ s, bf16_t* __restrict__ o, int R) {
    int row = blockIdx.x * WAVES_PER_BLOCK + (threadIdx.x >> 5);
    int lane = threadIdx.x & 31;
    if (row >= R) return;
    const float* p = s + (long)row * 512;
    bf16_t* q = o + (long)row * 512;
    float x[16];
    float mx = -3.4e38f;
    #pragma unroll
    for (int i = 0; i < 16; ++i) { x[i] = p[lane + i * 32]; mx = fmaxf(mx, x[i]); }
    mx = wave_max(mx);
    float sum = 0.0f;
    #pragma unroll
    for (int i = 0; i < 16; ++i) { x[i] = expf(x[i] - mx); sum += x[i]; }
    sum = wave_sum(sum);
    float inv = 1.0f / sum;
    #pragma unroll
    for (int i = 0; i < 16; ++i) q[lane + i * 32] = (bf16_t)(x[i] * inv);
}

// ---------------------------------------------------------------------------
// LayerNorm (768 wide, wave per row): f32 in -> bf16 out
// ---------------------------------------------------------------------------
__global__ __launch_bounds__(128) void layernorm768(
    const float* __restrict__ x, bf16_t* __restrict__ o,
    const float* __restrict__ g, const float* __restrict__ bb, int R) {
    int row = blockIdx.x * WAVES_PER_BLOCK + (threadIdx.x >> 5);
    int lane = threadIdx.x & 31;
    if (row >= R) return;
    const float* p = x + (long)row * 768;
    bf16_t* q = o + (long)row * 768;
    float v[24];
    float s = 0.0f;
    #pragma unroll
    for (int i = 0; i < 24; ++i) { v[i] = p[lane + i * 32]; s += v[i]; }
    s = wave_sum(s);
    const float mu = s * (1.0f / 768.0f);
    float qv = 0.0f;
    #pragma unroll
    for (int i = 0; i < 24; ++i) { float d = v[i] - mu; qv += d * d; }
    qv = wave_sum(qv);
    const float rstd = rsqrtf(qv * (1.0f / 768.0f) + 1e-12f);
    #pragma unroll
    for (int i = 0; i < 24; ++i) {
        int c = lane + i * 32;
        q[c] = (bf16_t)((v[i] - mu) * rstd * g[c] + bb[c]);
    }
}

// ---------------------------------------------------------------------------
extern "C" void kernel_launch(void* const* d_in, const int* in_sizes, int n_in,
                              void* d_out, int out_size, void* d_ws, size_t ws_size,
                              hipStream_t stream) {
    const float* src_scores = (const float*)d_in[0];  // [8,512,512]
    const float* seq_hidden = (const float*)d_in[1];  // [8,512,768]
    const float* seq_mask   = (const float*)d_in[2];  // [8,512]
    const float* pos_emb    = (const float*)d_in[3];  // [512,200]
    const float* Wp         = (const float*)d_in[4];  // [768,200]
    const float* bp         = (const float*)d_in[5];  // [768]
    const float* Wt         = (const float*)d_in[6];  // [768,768]
    const float* bt         = (const float*)d_in[7];  // [768]
    const float* ln_g       = (const float*)d_in[8];  // [768]
    const float* ln_b       = (const float*)d_in[9];  // [768]
    const float* Wd         = (const float*)d_in[10]; // [30522,768]
    const float* dec_bias   = (const float*)d_in[11]; // [30522]
    float* out = (float*)d_out;                       // [8,512,30522]

    // Workspace carve (all region sizes are 256B multiples)
    char* w = (char*)d_ws;
    bf16_t* pos_h_bf  = (bf16_t*)w; w += 512l * 768 * 2;
    bf16_t* seqN_bf   = (bf16_t*)w; w += 8l * 512 * 768 * 2;
    bf16_t* seqT_bf   = (bf16_t*)w; w += 8l * 768 * 512 * 2;
    bf16_t* Wt_bf     = (bf16_t*)w; w += 768l * 768 * 2;
    bf16_t* Wd_bf     = (bf16_t*)w; w += 30522l * 768 * 2;
    float*  scores    = (float*)w;               // dead after softmax ...
    float*  tg        = (float*)w; w += 8l * 512 * 768 * 4;  // ... region reused for tg
    bf16_t* alpha_bf  = (bf16_t*)w; w += 8l * 512 * 512 * 2;
    bf16_t* seqsum_bf = (bf16_t*)w; w += 8l * 512 * 768 * 2;
    bf16_t* tgb       = (bf16_t*)w; w += 8l * 512 * 768 * 2;

    // 0) one-time bf16 prep
    prep_seq_hidden<<<dim3(24, 16, 8), dim3(32, 8), 0, stream>>>(seq_hidden, seqN_bf, seqT_bf);
    cvt_bf16_kernel<<<(768l * 768 / 8 + 255) / 256, 256, 0, stream>>>(Wt, Wt_bf, 768l * 768);
    cvt_bf16_kernel<<<(30522l * 768 / 8 + 255) / 256, 256, 0, stream>>>(Wd, Wd_bf, 30522l * 768);

    // 1) pos_h = pos_emb @ Wp^T + bp  (bf16 out)
    posh_kernel<<<1536, 256, 0, stream>>>(pos_emb, Wp, bp, pos_h_bf);

    // 2) scores[b] = pos_h @ seq_hidden[b]^T + src + mask*(-1e4)  (M=N=512, K=768)
    //    tiles/batch = (512/32)*(512/64) = 128 -> 32 blocks
    gemm_bf16<EPI_SCORES, false><<<dim3(32, 8), 128, 0, stream>>>(
        pos_h_bf, 0, seqN_bf, 512l * 768, scores, 512l * 512,
        src_scores, 512l * 512, seq_mask, 512, 512, 512, 768);

    // 3) softmax (4096 rows of 512) -> bf16 alpha
    softmax512<<<1024, 128, 0, stream>>>(scores, alpha_bf, 4096);

    // 4) seq_sum[b] = alpha[b] @ seq_hidden[b]  via W = seq_hidden^T[h][s]
    //    (M=512,N=768,K=512): tiles/batch = 16*12 = 192 -> 48 blocks
    gemm_bf16<EPI_NONE, true><<<dim3(48, 8), 128, 0, stream>>>(
        alpha_bf, 512l * 512, seqT_bf, 768l * 512, seqsum_bf, 512l * 768,
        nullptr, 0, nullptr, 0, 512, 768, 512);

    // 5) tg = gelu(seq_sum @ Wt^T + bt)  (M=4096,N=768,K=768)
    //    tiles = 128*12 = 1536 -> 384 blocks
    gemm_bf16<EPI_GELU, false><<<dim3(384, 1), 128, 0, stream>>>(
        seqsum_bf, 0, Wt_bf, 0, tg, 0, bt, 0, nullptr, 0, 4096, 768, 768);

    // 6) LayerNorm -> bf16 h
    layernorm768<<<1024, 128, 0, stream>>>(tg, tgb, ln_g, ln_b, 4096);

    // 7) out = h @ Wd^T + dec_bias  (M=4096,N=30522,K=768)
    //    tiles = 128*477 = 61056 -> 15264 blocks
    gemm_bf16<EPI_BIAS, false><<<dim3(15264, 1), 128, 0, stream>>>(
        tgb, 0, Wd_bf, 0, out, 0, dec_bias, 0, nullptr, 0, 4096, 30522, 768);
}
